// MultiHeadAttention_ablatuon_21689584845397
// MI455X (gfx1250) — compile-verified, matrix operand order checked by reading, not run on hardware
//
#include <hip/hip_runtime.h>
#include <hip/hip_bf16.h>
#include <math.h>

// ---- problem constants (match reference) ----
#define B_ 2
#define N_ 2048
#define D_ 1024
#define H_ 16
#define DK_ 64
#define DV_ 64
#define M_ (B_ * N_)          // 4096 flattened rows
#define NEGV (-1000000000.0f)

// LDS pitches
#define SPITCH 2064   // f32 score strip pitch (floats)
#define PP_ 2056      // f16 P strip pitch (halves, 16B-aligned rows)

typedef __attribute__((ext_vector_type(16))) _Float16 v16h;
typedef __attribute__((ext_vector_type(8)))  _Float16 h8;
typedef __attribute__((ext_vector_type(4)))  _Float16 h4v;
typedef __attribute__((ext_vector_type(8)))  float    v8f;
typedef __attribute__((ext_vector_type(4)))  float    f4;

// CDNA5 wave32 WMMA 16x16x32 f16 fragment layout:
//  A (16x32): lane L row M=L%16; half=L/16:
//     elems 0..7  -> K = kb + half*8 + i      (contiguous -> b128)
//     elems 8..15 -> K = kb + 16 + half*8 + i (contiguous -> b128)
//  B (32x16): lane L col N=L%16; same K striping.
//  C/D f32: elem i -> row M = half*8 + i, col N = L%16.

__device__ __forceinline__ v16h cat16(h8 lo, h8 hi) {
  return __builtin_shufflevector(lo, hi, 0, 1, 2, 3, 4, 5, 6, 7,
                                 8, 9, 10, 11, 12, 13, 14, 15);
}

// ------------------------------------------------------------------
// f32 -> f16 elementwise convert (vectorized x4)
// ------------------------------------------------------------------
__global__ void cvt_f32_to_f16(const float* __restrict__ src,
                               _Float16* __restrict__ dst, int n4) {
  const int i = blockIdx.x * blockDim.x + threadIdx.x;
  if (i < n4) {
    const f4 v = ((const f4*)src)[i];
    ((h4v*)dst)[i] = __builtin_convertvector(v, h4v);
  }
}

// ------------------------------------------------------------------
// f32 -> f16 transpose-convert: WT[c][k] = (f16) W[k][c]
// block (16,16), grid (K/16, Ncols/16); LDS tile keeps both sides coalesced
// ------------------------------------------------------------------
__global__ void cvtT_f32_to_f16(const float* __restrict__ W,
                                _Float16* __restrict__ WT,
                                int K, int Ncols) {
  __shared__ float tile[16][17];
  const int k0 = blockIdx.x * 16, c0 = blockIdx.y * 16;
  const int tx = threadIdx.x, ty = threadIdx.y;
  tile[ty][tx] = W[(size_t)(k0 + ty) * Ncols + c0 + tx];
  __syncthreads();
  WT[(size_t)(c0 + ty) * K + k0 + tx] = (_Float16)tile[tx][ty];
}

// ------------------------------------------------------------------
// Shared GEMM core: one wave computes a 16x16 f32 tile of A16 @ WT^T.
// The 16x32 A block (shared by the block's 4 waves) is staged in LDS via
// CDNA5 async global->LDS copies (tracked by ASYNCcnt).
// A16: [M,K] f16 row-major;  WT: [Ncols,K] f16 (i.e. W transposed)
// ------------------------------------------------------------------
__device__ __forceinline__ v8f gemm16_core(const _Float16* __restrict__ A16,
                                           const _Float16* __restrict__ WT,
                                           _Float16* ldsA, int mt, int col,
                                           int K) {
  const int tid  = threadIdx.x;
  const int lane = tid & 31;
  const int half = lane >> 4;
  const int l16  = lane & 15;
  const _Float16* __restrict__ wrow = WT + (size_t)col * K;

  v8f acc = {};
  for (int kb = 0; kb < K; kb += 32) {
    __syncthreads();  // previous iteration's readers done
    if (tid < 64) {   // 16 rows x 4 chunks-of-16B (8 halves each)
      const int r  = tid >> 2;   // row 0..15
      const int ch = tid & 3;    // 16B chunk 0..3
      const _Float16* src = A16 + (size_t)(mt * 16 + r) * K + kb + ch * 8;
      const unsigned ldsOff =
          (unsigned)(size_t)(void*)(ldsA + r * 32 + ch * 8);
      const unsigned long long ga = (unsigned long long)(size_t)src;
      asm volatile("global_load_async_to_lds_b128 %0, %1, off"
                   :: "v"(ldsOff), "v"(ga)
                   : "memory");
    }
    asm volatile("s_wait_asynccnt 0x0" ::: "memory");
    __syncthreads();

    const h8 alo = *(const h8*)(ldsA + l16 * 32 + half * 8);
    const h8 ahi = *(const h8*)(ldsA + l16 * 32 + 16 + half * 8);
    const h8 blo = *(const h8*)(wrow + kb + half * 8);
    const h8 bhi = *(const h8*)(wrow + kb + 16 + half * 8);
    __builtin_prefetch(wrow + kb + 64, 0, 0);  // global_prefetch_b8

    acc = __builtin_amdgcn_wmma_f32_16x16x32_f16(
        false, cat16(alo, ahi), false, cat16(blo, bhi),
        (short)0, acc, false, false);
  }
  return acc;
}

// ------------------------------------------------------------------
// Projection GEMM: C[M,1024] = A16 @ W (via WT), f16 out.
// storeVT==0: row-major [M,1024]. storeVT==1: per-head transposed
// layout [b][h][dv][N] (used for V so P@V fragments are contiguous).
// grid (M/16, 1024/64), block 128 (4 waves)
// ------------------------------------------------------------------
__global__ void mha_proj_gemm(const _Float16* __restrict__ A16,
                              const _Float16* __restrict__ WT,
                              _Float16* __restrict__ C, int storeVT) {
  __shared__ __attribute__((aligned(16))) _Float16 ldsA[16 * 32];
  const int wave = threadIdx.x >> 5;
  const int lane = threadIdx.x & 31;
  const int half = lane >> 4;
  const int l16  = lane & 15;
  const int mt  = blockIdx.x;
  const int col = (blockIdx.y * 4 + wave) * 16 + l16;

  const v8f acc = gemm16_core(A16, WT, ldsA, mt, col, D_);

  if (storeVT == 0) {
#pragma unroll
    for (int i = 0; i < 8; ++i) {
      const int m = mt * 16 + half * 8 + i;
      C[((size_t)m << 10) + col] = (_Float16)acc[i];
    }
  } else {
#pragma unroll
    for (int i = 0; i < 8; ++i) {
      const int m = mt * 16 + half * 8 + i;
      const int b = m >> 11, nseq = m & 2047;
      const int hh = col >> 6, d = col & 63;
      C[(((size_t)((b * H_ + hh) * DV_ + d)) << 11) + nseq] = (_Float16)acc[i];
    }
  }
}

// ------------------------------------------------------------------
// Attention: one block (8 wave32, ~207KB LDS) per (b, h, 16-row q tile)
// ------------------------------------------------------------------
__global__ void mha_attn_kernel(const _Float16* __restrict__ qh,
                                const _Float16* __restrict__ kh,
                                const _Float16* __restrict__ vhT,
                                const int* __restrict__ mask,
                                float* __restrict__ attn_out,
                                _Float16* __restrict__ ctx) {
  extern __shared__ char smem[];
  char* p = smem;
  float* S = (float*)p;            p += (size_t)16 * SPITCH * sizeof(float);
  _Float16* P16 = (_Float16*)p;    p += (size_t)16 * PP_ * sizeof(_Float16);
  float* red = (float*)p;          p += 256 * sizeof(float);
  float* rowmax = (float*)p;       p += 16 * sizeof(float);
  float* rowsum = (float*)p;       p += 16 * sizeof(float);
  float* opart = (float*)p;        // 2*16*64 floats

  const int tid  = threadIdx.x;
  const int wave = tid >> 5;
  const int lane = tid & 31;
  const int half = lane >> 4;
  const int l16  = lane & 15;
  const int qt = blockIdx.x, h = blockIdx.y, b = blockIdx.z;
  const int growbase = qt * 16;

  // ---- Q fragments once, scaled by 1/sqrt(DK)=0.125 (exact in f16) ----
  const _Float16* __restrict__ qrow =
      qh + (((size_t)(b * N_ + growbase + l16)) << 10) + (h << 6);
  h8 q0lo = *(const h8*)(qrow + half * 8);
  h8 q0hi = *(const h8*)(qrow + 16 + half * 8);
  h8 q1lo = *(const h8*)(qrow + 32 + half * 8);
  h8 q1hi = *(const h8*)(qrow + 48 + half * 8);
  const _Float16 sc = (_Float16)0.125f;
  q0lo *= sc; q0hi *= sc; q1lo *= sc; q1hi *= sc;
  const v16h a0 = cat16(q0lo, q0hi);
  const v16h a1 = cat16(q1lo, q1hi);

  // ---- phase 1: masked scores -> LDS (wave w owns 16 column tiles) ----
  for (int ct = wave * 16; ct < wave * 16 + 16; ++ct) {
    const int c0 = ct * 16;
    const _Float16* __restrict__ krow =
        kh + (((size_t)(b * N_ + c0 + l16)) << 10) + (h << 6);
    const h8 k0lo = *(const h8*)(krow + half * 8);
    const h8 k0hi = *(const h8*)(krow + 16 + half * 8);
    const h8 k1lo = *(const h8*)(krow + 32 + half * 8);
    const h8 k1hi = *(const h8*)(krow + 48 + half * 8);
    v8f acc = {};
    acc = __builtin_amdgcn_wmma_f32_16x16x32_f16(
        false, a0, false, cat16(k0lo, k0hi), (short)0, acc, false, false);
    acc = __builtin_amdgcn_wmma_f32_16x16x32_f16(
        false, a1, false, cat16(k1lo, k1hi), (short)0, acc, false, false);
    const int gcol = c0 + l16;
#pragma unroll
    for (int i = 0; i < 8; ++i) {
      const int m = half * 8 + i;
      const int grow = growbase + m;
      float val = acc[i];
      const int mk = mask[(((size_t)(b * N_ + grow)) << 11) + gcol];
      if (mk == 0 || gcol > grow) val = NEGV;  // external + causal mask
      S[m * SPITCH + gcol] = val;
    }
  }
  __syncthreads();

  // ---- phase 2: row max ----
  {
    const int r = tid >> 4, seg = tid & 15;
    float mx = -3.4e38f;
    const int base = seg * 128;
    for (int j = 0; j < 128; ++j) mx = fmaxf(mx, S[r * SPITCH + base + j]);
    red[r * 16 + seg] = mx;
  }
  __syncthreads();
  if (tid < 16) {
    float mx = -3.4e38f;
    for (int j = 0; j < 16; ++j) mx = fmaxf(mx, red[tid * 16 + j]);
    rowmax[tid] = mx;
  }
  __syncthreads();
  // ---- row sum of exp ----
  {
    const int r = tid >> 4, seg = tid & 15;
    const float mx = rowmax[r];
    float sm = 0.f;
    const int base = seg * 128;
    for (int j = 0; j < 128; ++j) sm += __expf(S[r * SPITCH + base + j] - mx);
    red[r * 16 + seg] = sm;
  }
  __syncthreads();
  if (tid < 16) {
    float sm = 0.f;
    for (int j = 0; j < 16; ++j) sm += red[tid * 16 + j];
    rowsum[tid] = sm;
  }
  __syncthreads();

  // ---- normalize: write attn to HBM once + f16 P strip to LDS ----
  float* __restrict__ attn_base =
      attn_out + (((size_t)((b * H_ + h) * N_ + growbase)) << 11);
  for (int j = 0; j < 128; ++j) {
    const int e = tid + j * 256;  // 16*2048 elements
    const int r = e >> 11;
    const int c = e & 2047;
    const float pv = __expf(S[r * SPITCH + c] - rowmax[r]) / rowsum[r];
    P16[r * PP_ + c] = (_Float16)pv;
    attn_base[((size_t)r << 11) + c] = pv;
  }
  __syncthreads();

  // ---- phase 3: ctx = P @ V (split-K over waves), all-b128 fragments ----
  const int ct    = wave & 3;   // 16-col tile of DV
  const int khalf = wave >> 2;  // K half
  const _Float16* __restrict__ vrow =
      vhT + (((size_t)((b * H_ + h) * DV_ + ct * 16 + l16)) << 11);
  v8f acc = {};
  for (int kb = khalf * 1024; kb < khalf * 1024 + 1024; kb += 32) {
    const h8 alo = *(const h8*)(P16 + l16 * PP_ + kb + half * 8);
    const h8 ahi = *(const h8*)(P16 + l16 * PP_ + kb + 16 + half * 8);
    const h8 blo = *(const h8*)(vrow + kb + half * 8);
    const h8 bhi = *(const h8*)(vrow + kb + 16 + half * 8);
    acc = __builtin_amdgcn_wmma_f32_16x16x32_f16(
        false, cat16(alo, ahi), false, cat16(blo, bhi),
        (short)0, acc, false, false);
  }
#pragma unroll
  for (int i = 0; i < 8; ++i) {
    const int m = half * 8 + i;
    opart[(khalf * 16 + m) * 64 + ct * 16 + l16] = acc[i];
  }
  __syncthreads();
  // combine split-K halves, store ctx (f16, row-major [M,1024])
  for (int j = 0; j < 4; ++j) {
    const int e = tid + j * 256;  // 16*64 elements
    const int m = e >> 6;
    const int c = e & 63;
    const float v = opart[m * 64 + c] + opart[(16 + m) * 64 + c];
    ctx[(((size_t)(b * N_ + growbase + m)) << 10) + (h << 6) + c] =
        (_Float16)v;
  }
}

// ------------------------------------------------------------------
// Output projection + residual: o_pre = ctx @ Wo + q   (f32 out)
// ------------------------------------------------------------------
__global__ void mha_out_gemm(const _Float16* __restrict__ ctx,
                             const _Float16* __restrict__ WoT,
                             const float* __restrict__ res,
                             float* __restrict__ C) {
  __shared__ __attribute__((aligned(16))) _Float16 ldsA[16 * 32];
  const int wave = threadIdx.x >> 5;
  const int lane = threadIdx.x & 31;
  const int half = lane >> 4;
  const int l16  = lane & 15;
  const int mt  = blockIdx.x;
  const int col = (blockIdx.y * 4 + wave) * 16 + l16;

  const v8f acc = gemm16_core(ctx, WoT, ldsA, mt, col, H_ * DV_);

#pragma unroll
  for (int i = 0; i < 8; ++i) {
    const int m = mt * 16 + half * 8 + i;
    C[((size_t)m << 10) + col] = acc[i] + res[((size_t)m << 10) + col];
  }
}

// ------------------------------------------------------------------
// LayerNorm per row of 1024 (eps=1e-6)
// ------------------------------------------------------------------
__global__ void mha_layernorm(const float* __restrict__ x,
                              const float* __restrict__ gamma,
                              const float* __restrict__ beta,
                              float* __restrict__ out) {
  __shared__ float red[256];
  const int row = blockIdx.x;
  const int tid = threadIdx.x;
  const float* __restrict__ xr = x + ((size_t)row << 10);

  float s = 0.f;
  for (int j = 0; j < 4; ++j) s += xr[tid + j * 256];
  red[tid] = s;
  __syncthreads();
  for (int off = 128; off > 0; off >>= 1) {
    if (tid < off) red[tid] += red[tid + off];
    __syncthreads();
  }
  const float mean = red[0] * (1.f / D_);
  __syncthreads();

  float v = 0.f;
  for (int j = 0; j < 4; ++j) {
    const float d = xr[tid + j * 256] - mean;
    v += d * d;
  }
  red[tid] = v;
  __syncthreads();
  for (int off = 128; off > 0; off >>= 1) {
    if (tid < off) red[tid] += red[tid + off];
    __syncthreads();
  }
  const float rstd = rsqrtf(red[0] * (1.f / D_) + 1e-6f);

  for (int j = 0; j < 4; ++j) {
    const int c = tid + j * 256;
    out[((size_t)row << 10) + c] = (xr[c] - mean) * rstd * gamma[c] + beta[c];
  }
}

// ------------------------------------------------------------------
extern "C" void kernel_launch(void* const* d_in, const int* in_sizes, int n_in,
                              void* d_out, int out_size, void* d_ws, size_t ws_size,
                              hipStream_t stream) {
  const float* q     = (const float*)d_in[0];
  const float* k     = (const float*)d_in[1];
  const float* v     = (const float*)d_in[2];
  const int*   mask  = (const int*)d_in[3];
  const float* Wq    = (const float*)d_in[4];
  const float* Wk    = (const float*)d_in[5];
  const float* Wv    = (const float*)d_in[6];
  const float* Wo    = (const float*)d_in[7];
  const float* gamma = (const float*)d_in[8];
  const float* beta  = (const float*)d_in[9];

  float* out  = (float*)d_out;                  // [B,N,D]
  float* attn = out + (size_t)B_ * N_ * D_;     // [B,H,N,N]

  // workspace carve-up (~84 MB)
  char* ws = (char*)d_ws;
  const size_t MD = (size_t)M_ * D_;
  _Float16* q16 = (_Float16*)ws;  ws += MD * sizeof(_Float16);
  _Float16* k16 = (_Float16*)ws;  ws += MD * sizeof(_Float16);
  _Float16* v16 = (_Float16*)ws;  ws += MD * sizeof(_Float16);
  _Float16* WqT = (_Float16*)ws;  ws += (size_t)D_ * D_ * sizeof(_Float16);
  _Float16* WkT = (_Float16*)ws;  ws += (size_t)D_ * D_ * sizeof(_Float16);
  _Float16* WvT = (_Float16*)ws;  ws += (size_t)D_ * D_ * sizeof(_Float16);
  _Float16* WoT = (_Float16*)ws;  ws += (size_t)D_ * D_ * sizeof(_Float16);
  _Float16* qh  = (_Float16*)ws;  ws += MD * sizeof(_Float16);
  _Float16* kh  = (_Float16*)ws;  ws += MD * sizeof(_Float16);
  _Float16* vhT = (_Float16*)ws;  ws += MD * sizeof(_Float16);
  _Float16* ctx = (_Float16*)ws;  ws += MD * sizeof(_Float16);
  float* o_pre  = (float*)ws;     ws += MD * sizeof(float);

  // 0) precision/layout prep
  const int n4 = (int)(MD / 4);
  cvt_f32_to_f16<<<n4 / 256, 256, 0, stream>>>(q, q16, n4);
  cvt_f32_to_f16<<<n4 / 256, 256, 0, stream>>>(k, k16, n4);
  cvt_f32_to_f16<<<n4 / 256, 256, 0, stream>>>(v, v16, n4);
  dim3 tGrid(D_ / 16, D_ / 16), tBlock(16, 16);
  cvtT_f32_to_f16<<<tGrid, tBlock, 0, stream>>>(Wq, WqT, D_, D_);
  cvtT_f32_to_f16<<<tGrid, tBlock, 0, stream>>>(Wk, WkT, D_, D_);
  cvtT_f32_to_f16<<<tGrid, tBlock, 0, stream>>>(Wv, WvT, D_, D_);
  cvtT_f32_to_f16<<<tGrid, tBlock, 0, stream>>>(Wo, WoT, D_, D_);

  // 1) QKV projections (V stored transposed per head)
  dim3 gGrid(M_ / 16, D_ / 64), gBlock(128);
  mha_proj_gemm<<<gGrid, gBlock, 0, stream>>>(q16, WqT, qh, 0);
  mha_proj_gemm<<<gGrid, gBlock, 0, stream>>>(k16, WkT, kh, 0);
  mha_proj_gemm<<<gGrid, gBlock, 0, stream>>>(v16, WvT, vhT, 1);

  // 2) attention (dynamic LDS ~207 KB; CDNA5 WGP has 320 KB)
  const size_t smemBytes = (size_t)16 * SPITCH * sizeof(float) +
                           (size_t)16 * PP_ * sizeof(_Float16) +
                           (256 + 16 + 16 + 2 * 16 * 64) * sizeof(float);
  mha_attn_kernel<<<dim3(N_ / 16, H_, B_), 256, smemBytes, stream>>>(
      qh, kh, vhT, mask, attn, ctx);

  // 3) output projection + residual
  mha_out_gemm<<<gGrid, gBlock, 0, stream>>>(ctx, WoT, q, o_pre);

  // 4) LayerNorm
  mha_layernorm<<<M_, 256, 0, stream>>>(o_pre, gamma, beta, out);
}
